// SASA_43267500540812
// MI455X (gfx1250) — compile-verified, matrix operand order checked by reading, not run on hardware
//
#include <hip/hip_runtime.h>
#include <math.h>

// ---------------------------------------------------------------------------
// SASA attention block, fused for gfx1250 (MI455X).
//   B=4, C=256, H=W=64 -> N=4096, Cq=32.
//   mean(att) computed algebraically: sum att = sum_q (sum_n Q)(sum_m K).
//   Main kernel: flash-style V @ sigmoid(QK^T - mean)^T with bf16 WMMA.
// ---------------------------------------------------------------------------

#define BATCH 4
#define C_IN  256
#define CQ    32
#define N_SP  4096
#define NBLK  64          // n-columns per projection workgroup
#define NCHUNK 32         // K-dim chunk of the fused attention loop

typedef __attribute__((ext_vector_type(16))) __bf16  v16bf;
typedef __attribute__((ext_vector_type(8)))  float   v8f;
typedef __attribute__((ext_vector_type(4)))  float   v4f;
typedef __attribute__((ext_vector_type(4)))  unsigned int v4u;

union FragU { v16bf v; v4u q[2]; };

static __device__ inline unsigned short f32_to_bf16(float f) {
  unsigned int u = __float_as_uint(f);
  u += 0x7FFFu + ((u >> 16) & 1u);     // round to nearest even
  return (unsigned short)(u >> 16);
}

// Fast sigmoid: result is rounded to bf16 anyway, so v_rcp_f32 (~1 ulp) is
// plenty; avoids the IEEE div_scale/div_fmas Newton chain.
static __device__ inline float fast_sigmoid(float x) {
  return __builtin_amdgcn_rcpf(1.0f + __expf(-x));
}

// Gather one 16x32 (bf16) WMMA A/B fragment from a row-major [row][32] array.
// A-matrix: row = M (lane&15); B-matrix: row = column n (lane&15).
// K layout per ISA: lanes 0-15 hold K {0-7,16-23}, lanes 16-31 hold K {8-15,24-31}.
static __device__ inline v16bf load_frag32(const unsigned short* rowbase, int lane) {
  FragU f;
  const unsigned short* p = rowbase + ((lane >> 4) << 3);  // +0 or +8 halves
  f.q[0] = *reinterpret_cast<const v4u*>(p);
  f.q[1] = *reinterpret_cast<const v4u*>(p + 16);
  return f.v;
}

// ---------------------------------------------------------------------------
// Kernel 1: projections. Per block: one batch b, one 64-wide n slab.
//   Phase 1: each thread owns one V output channel (vi = tid), 64 n.
//   Phase 2: each thread owns one q/k row (tid&63) over a 16-n quarter.
// Writes Qbf[b][n][32], Ktbf[b][n][32], Vbf[b][c][4096] (bf16) and
// deterministic per-block partial sums of Q and K (for the global mean).
// ---------------------------------------------------------------------------
__global__ __launch_bounds__(256) void sasa_proj_kernel(
    const float* __restrict__ X,
    const float* __restrict__ Wq, const float* __restrict__ bq,
    const float* __restrict__ Wk, const float* __restrict__ bk,
    const float* __restrict__ Wv, const float* __restrict__ bv,
    unsigned short* __restrict__ Qbf,
    unsigned short* __restrict__ Ktbf,
    unsigned short* __restrict__ Vbf,
    float* __restrict__ qpart,   // [B][64 blocks][32]
    float* __restrict__ kpart)   // [B][64 blocks][32]
{
  __shared__ float Xs[C_IN][NBLK];     // 64 KB
  __shared__ float part[64][4];        // q/k row partial sums per quarter

  const int b   = blockIdx.y;
  const int blk = blockIdx.x;
  const int n0  = blk * NBLK;
  const int tid = threadIdx.x;

  const float* Xb = X + (size_t)b * C_IN * N_SP;
  for (int idx = tid; idx < C_IN * NBLK; idx += 256) {
    Xs[idx >> 6][idx & 63] = Xb[(size_t)(idx >> 6) * N_SP + n0 + (idx & 63)];
  }
  __syncthreads();

  // ---- Phase 1: V row (channel vi = tid) across all 64 n ----
  {
    float acc[NBLK];
    const float bias = bv[tid];
#pragma unroll
    for (int n = 0; n < NBLK; ++n) acc[n] = bias;
    const float* w = Wv + tid * C_IN;
    for (int c = 0; c < C_IN; ++c) {
      const float wc = w[c];
      const v4f* xr = reinterpret_cast<const v4f*>(&Xs[c][0]);
#pragma unroll
      for (int n4 = 0; n4 < NBLK / 4; ++n4) {
        v4f x = xr[n4];
        acc[4*n4+0] = fmaf(wc, x.x, acc[4*n4+0]);
        acc[4*n4+1] = fmaf(wc, x.y, acc[4*n4+1]);
        acc[4*n4+2] = fmaf(wc, x.z, acc[4*n4+2]);
        acc[4*n4+3] = fmaf(wc, x.w, acc[4*n4+3]);
      }
    }
    unsigned short* vrow = Vbf + (size_t)(b * C_IN + tid) * N_SP + n0;
#pragma unroll
    for (int n8 = 0; n8 < 8; ++n8) {
      v4u u;
      u.x = (unsigned)f32_to_bf16(acc[8*n8+0]) | ((unsigned)f32_to_bf16(acc[8*n8+1]) << 16);
      u.y = (unsigned)f32_to_bf16(acc[8*n8+2]) | ((unsigned)f32_to_bf16(acc[8*n8+3]) << 16);
      u.z = (unsigned)f32_to_bf16(acc[8*n8+4]) | ((unsigned)f32_to_bf16(acc[8*n8+5]) << 16);
      u.w = (unsigned)f32_to_bf16(acc[8*n8+6]) | ((unsigned)f32_to_bf16(acc[8*n8+7]) << 16);
      *reinterpret_cast<v4u*>(vrow + 8 * n8) = u;
    }
  }

  // ---- Phase 2: q/k rows. row<32 -> Q row; row>=32 -> K row. 16-n quarter. ----
  {
    const int row = tid & 63;
    const int qtr = tid >> 6;
    const int nb  = qtr * 16;
    const bool isq = (row < 32);
    const int r32 = isq ? row : (row - 32);
    const float* w   = (isq ? Wq : Wk) + r32 * C_IN;
    const float bias = isq ? bq[r32] : bk[r32];

    float acc[16];
#pragma unroll
    for (int n = 0; n < 16; ++n) acc[n] = bias;
    for (int c = 0; c < C_IN; ++c) {
      const float wc = w[c];
      const v4f* xr = reinterpret_cast<const v4f*>(&Xs[c][nb]);
#pragma unroll
      for (int n4 = 0; n4 < 4; ++n4) {
        v4f x = xr[n4];
        acc[4*n4+0] = fmaf(wc, x.x, acc[4*n4+0]);
        acc[4*n4+1] = fmaf(wc, x.y, acc[4*n4+1]);
        acc[4*n4+2] = fmaf(wc, x.z, acc[4*n4+2]);
        acc[4*n4+3] = fmaf(wc, x.w, acc[4*n4+3]);
      }
    }
    unsigned short* dst = isq ? Qbf : Ktbf;
    float s = 0.f;
#pragma unroll
    for (int n = 0; n < 16; ++n) {
      s += acc[n];
      dst[(size_t)(b * N_SP + n0 + nb + n) * CQ + r32] = f32_to_bf16(acc[n]);
    }
    part[row][qtr] = s;
  }
  __syncthreads();

  if (tid < 64) {
    const float t = part[tid][0] + part[tid][1] + part[tid][2] + part[tid][3];
    if (tid < 32) qpart[(size_t)(b * 64 + blk) * 32 + tid]        = t;
    else          kpart[(size_t)(b * 64 + blk) * 32 + (tid - 32)] = t;
  }
}

// ---------------------------------------------------------------------------
// Kernel 2: global mean of att via sum_b sum_q (sum_n Q)(sum_m K) / (B*N*N).
// Fully deterministic tree reduction.
// ---------------------------------------------------------------------------
__global__ __launch_bounds__(128) void sasa_mean_kernel(
    const float* __restrict__ qpart, const float* __restrict__ kpart,
    float* __restrict__ meanp)
{
  __shared__ float red[128];
  const int t = threadIdx.x;           // b = t>>5, q = t&31
  const int b = t >> 5, q = t & 31;
  float sq = 0.f, sk = 0.f;
  for (int blk = 0; blk < 64; ++blk) {
    sq += qpart[(size_t)(b * 64 + blk) * 32 + q];
    sk += kpart[(size_t)(b * 64 + blk) * 32 + q];
  }
  red[t] = sq * sk;
  __syncthreads();
  for (int s = 64; s > 0; s >>= 1) {
    if (t < s) red[t] += red[t + s];
    __syncthreads();
  }
  if (t == 0)
    meanp[0] = red[0] * (1.0f / ((float)BATCH * (float)N_SP * (float)N_SP));
}

// ---------------------------------------------------------------------------
// Kernel 3: fused  out[b,c,m] = gamma * sum_n V[c,n] * sigmoid(Q[m].K[n]-mean) + X
// Block: one batch b, 64 m columns, all 256 channels. 8 waves:
//   score duty : wave computes one 16m x 16n att tile per 32-n chunk
//                (single v_wmma_f32_16x16x32_bf16, Cq==32==K), sigmoids it,
//                scatters bf16 P into LDS already in B-fragment layout.
//   output duty: wave owns c-tiles {2w,2w+1} x m-tiles {0..3}: 8 WMMAs/chunk.
// P double-buffered -> one barrier per chunk.
// ---------------------------------------------------------------------------
__global__ __launch_bounds__(256) void sasa_attn_kernel(
    const unsigned short* __restrict__ Qbf,
    const unsigned short* __restrict__ Ktbf,
    const unsigned short* __restrict__ Vbf,
    const float* __restrict__ meanp,
    const float* __restrict__ X,
    const float* __restrict__ gammap,
    float* __restrict__ out)
{
  __shared__ unsigned short Plds[2 * 4 * 32 * 16];  // 2 bufs * 4 m-tiles * 32 lanes * 16 halves = 8 KB

  const int b    = blockIdx.y;
  const int m0   = blockIdx.x * 64;
  const int tid  = threadIdx.x;
  const int wave = tid >> 5;
  const int lane = tid & 31;

  const float mean  = meanp[0];
  const float gamma = gammap[0];

  const int mt_s = wave & 3;   // score m-tile of this wave
  const int ns_s = wave >> 2;  // score n-subtile (0/1) of this wave

  const unsigned short* Qb = Qbf  + (size_t)(b * N_SP) * CQ;
  const unsigned short* Kb = Ktbf + (size_t)(b * N_SP) * CQ;
  const unsigned short* Vb = Vbf  + (size_t)(b * C_IN) * N_SP;

  // Persistent A fragment: Q rows of this wave's score m-tile (16 x 32 bf16).
  const v16bf qfrag =
      load_frag32(Qb + (size_t)(m0 + mt_s * 16 + (lane & 15)) * CQ, lane);

  v8f acc[2][4];
  const v8f vzero = {0.f, 0.f, 0.f, 0.f, 0.f, 0.f, 0.f, 0.f};
#pragma unroll
  for (int i = 0; i < 2; ++i)
#pragma unroll
    for (int j = 0; j < 4; ++j) acc[i][j] = vzero;

  // P scatter mapping (score accumulator -> B-fragment layout), see ISA 7.12.2:
  //   att elem (lane,r): m_local = r + 8*(lane>=16), K index n_k = 16*ns_s + (lane&15)
  //   dest lane L = m_local + 16*bit3(n_k) ; dest elem e = (n_k&7) + 8*(n_k>>4)
  const int p_e     = (lane & 7) + 8 * ns_s;
  const int p_lbase = 16 * ((lane >> 3) & 1) + 8 * (lane >> 4);

  const int ct0 = wave * 2;
  const int ct1 = wave * 2 + 1;

  for (int k = 0; k < N_SP / NCHUNK; ++k) {
    const int n0  = k * NCHUNK;
    const int buf = k & 1;

    // Issue all global loads for this chunk up front so their latency overlaps
    // the score WMMA, the sigmoid transcendentals, and the barrier.
    const v16bf kfrag =
        load_frag32(Kb + (size_t)(n0 + ns_s * 16 + (lane & 15)) * CQ, lane);
    const v16bf vf0 =
        load_frag32(Vb + (size_t)(ct0 * 16 + (lane & 15)) * N_SP + n0, lane);
    const v16bf vf1 =
        load_frag32(Vb + (size_t)(ct1 * 16 + (lane & 15)) * N_SP + n0, lane);

    // ---- scores: att[m_tile, 16n] = Q(16x32) x K^T(32x16) ----
    v8f att = __builtin_amdgcn_wmma_f32_16x16x32_bf16(
        false, qfrag, false, kfrag, (short)0, vzero, false, false);

    unsigned short* pb = Plds + (size_t)(buf * 4 + mt_s) * 32 * 16;
#pragma unroll
    for (int r = 0; r < 8; ++r) {
      const float s = fast_sigmoid(att[r] - mean);   // sigmoid(att - mean)
      pb[(r + p_lbase) * 16 + p_e] = f32_to_bf16(s);
    }
    __syncthreads();

    // ---- output: acc[c_tile, m_tile] += V(16c x 32n) x P(32n x 16m) ----
    const unsigned short* plb = Plds + (size_t)(buf * 4) * 32 * 16;
    v16bf pfrag[4];
#pragma unroll
    for (int mt = 0; mt < 4; ++mt) {
      FragU f;
      const v4u* p = reinterpret_cast<const v4u*>(plb + (mt * 32 + lane) * 16);
      f.q[0] = p[0];
      f.q[1] = p[1];
      pfrag[mt] = f.v;
    }
#pragma unroll
    for (int mt = 0; mt < 4; ++mt) {
      acc[0][mt] = __builtin_amdgcn_wmma_f32_16x16x32_bf16(
          false, vf0, false, pfrag[mt], (short)0, acc[0][mt], false, false);
      acc[1][mt] = __builtin_amdgcn_wmma_f32_16x16x32_bf16(
          false, vf1, false, pfrag[mt], (short)0, acc[1][mt], false, false);
    }
  }

  // ---- epilogue: out = gamma*acc + X ----
  const float* Xb = X   + (size_t)(b * C_IN) * N_SP;
  float*       Ob = out + (size_t)(b * C_IN) * N_SP;
#pragma unroll
  for (int i = 0; i < 2; ++i) {
    const int ct = wave * 2 + i;
#pragma unroll
    for (int mt = 0; mt < 4; ++mt) {
      const int m = m0 + mt * 16 + (lane & 15);
#pragma unroll
      for (int r = 0; r < 8; ++r) {
        const int c = ct * 16 + r + 8 * (lane >> 4);
        const size_t idx = (size_t)c * N_SP + m;
        Ob[idx] = fmaf(gamma, acc[i][mt][r], Xb[idx]);
      }
    }
  }
}

// ---------------------------------------------------------------------------
// Workspace layout (bytes):
//   0        : mean (float), padded
//   4096     : qpart  [4*64*32] f32   (32 KB)
//   36864    : kpart  [4*64*32] f32   (32 KB)
//   69632    : Qbf    [4*4096*32] bf16 (1 MB)
//   1118208  : Ktbf   [4*4096*32] bf16 (1 MB)
//   2166784  : Vbf    [4*256*4096] bf16 (8 MB)
//   total ~= 10.6 MB
// ---------------------------------------------------------------------------
extern "C" void kernel_launch(void* const* d_in, const int* in_sizes, int n_in,
                              void* d_out, int out_size, void* d_ws, size_t ws_size,
                              hipStream_t stream) {
  const float* X     = (const float*)d_in[0];
  const float* PAF   = (const float*)d_in[1];
  const float* Wq    = (const float*)d_in[2];
  const float* bq    = (const float*)d_in[3];
  const float* Wk    = (const float*)d_in[4];
  const float* bk    = (const float*)d_in[5];
  const float* Wv    = (const float*)d_in[6];
  const float* bv    = (const float*)d_in[7];
  const float* gamma = (const float*)d_in[8];
  float* out = (float*)d_out;

  char* ws = (char*)d_ws;
  float* meanp = (float*)ws;
  float* qpart = (float*)(ws + 4096);
  float* kpart = (float*)(ws + 36864);
  unsigned short* Qbf  = (unsigned short*)(ws + 69632);
  unsigned short* Ktbf = (unsigned short*)(ws + 1118208);
  unsigned short* Vbf  = (unsigned short*)(ws + 2166784);

  dim3 gproj(N_SP / NBLK, BATCH);   // 64 x 4
  sasa_proj_kernel<<<gproj, 256, 0, stream>>>(X, Wq, bq, Wk, bk, Wv, bv,
                                              Qbf, Ktbf, Vbf, qpart, kpart);

  sasa_mean_kernel<<<1, 128, 0, stream>>>(qpart, kpart, meanp);

  dim3 gattn(N_SP / 64, BATCH);     // 64 x 4
  sasa_attn_kernel<<<gattn, 256, 0, stream>>>(Qbf, Ktbf, Vbf, meanp, X, gamma, out);

  // second tuple output: PAF_mag passthrough
  const size_t half = (size_t)BATCH * C_IN * N_SP;
  hipMemcpyAsync(out + half, PAF, half * sizeof(float),
                 hipMemcpyDeviceToDevice, stream);
}
